// RecurrentDecoderTrajectoryTensor_42494406426985
// MI455X (gfx1250) — compile-verified, matrix operand order checked by reading, not run on hardware
//
#include <hip/hip_runtime.h>
#include <hip/hip_bf16.h>

typedef __bf16 bf16_t;
typedef __attribute__((ext_vector_type(16))) __bf16 v16bf;
typedef __attribute__((ext_vector_type(8)))  __bf16 v8bf;
typedef __attribute__((ext_vector_type(4)))  __bf16 v4bf;
typedef __attribute__((ext_vector_type(8)))  float  v8f;

// ---------------------------------------------------------------------------
// Build a 16x32 bf16 WMMA fragment (A-style striping, ISA 7.12.2) from a
// row-major row at `p` (already offset to this lane's row + kOff elements).
// ---------------------------------------------------------------------------
__device__ __forceinline__ v16bf frag_from_row(const bf16_t* p) {
    v8bf lo = *(const v8bf*)(p);
    v8bf hi = *(const v8bf*)(p + 16);
    return __builtin_shufflevector(lo, hi, 0,1,2,3,4,5,6,7,8,9,10,11,12,13,14,15);
}

__device__ __forceinline__ float sigmoidf_fast(float x) {
    return 1.0f / (1.0f + __expf(-x));
}

__device__ __forceinline__ float tanhf_fast(float x) {
    x = fminf(fmaxf(x, -15.0f), 15.0f);           // avoid inf/inf
    float e = __expf(2.0f * x);
    return (e - 1.0f) / (e + 1.0f);
}

// ---------------------------------------------------------------------------
// f32 -> bf16 convert, 4 elements/thread (all operand sizes divisible by 4)
// ---------------------------------------------------------------------------
__global__ void f32_to_bf16_kernel(const float* __restrict__ in,
                                   bf16_t* __restrict__ out, size_t n4) {
    size_t i = (size_t)blockIdx.x * blockDim.x + threadIdx.x;
    if (i >= n4) return;
    const float4 v = ((const float4*)in)[i];
    v4bf o = { (bf16_t)v.x, (bf16_t)v.y, (bf16_t)v.z, (bf16_t)v.w };
    ((v4bf*)out)[i] = o;
}

// ---------------------------------------------------------------------------
// Workgroup-cooperative bf16 GEMM with LDS tiling + double buffering.
//   C[m][n] = epilogue( sum_k A[m][k] * W[n][k] + bias[n] )
// Block: 128 threads = 4 waves. __launch_bounds__(128, 1) relaxes the
// backend's occupancy-driven VGPR cap so the double-buffer staging registers
// are NOT spilled to scratch (observed with default bounds).
// Block tile 64(M) x 64(N); wave w computes 64 x 16 at N offset w*16 with
// 4 accumulators. All waves share the A fragments (LDS broadcast).
// K-step 64, double-buffered. LDS rows padded to 144B: ds_load_b128 start
// bank group = (9*lane) mod 16, distinct across lanes 0..15 -> conflict-free.
// Requires M%64==0, N%64==0, K%64==0 (true for this problem).
// ---------------------------------------------------------------------------
#define LDS_STRIDE 72   // elements per LDS row (64 data + 8 pad) = 144 bytes

template <bool APPLY_SIGMOID>
__global__ __launch_bounds__(128, 1)
void gemm_bf16_lds_kernel(const bf16_t* __restrict__ A,    // [M, K] bf16
                          const bf16_t* __restrict__ W,    // [N, K] bf16
                          const float*  __restrict__ bias, // [N]
                          float* __restrict__ C,           // [M, N] f32
                          int M, int K, int N, int tilesN /* N/64 */) {
    __shared__ bf16_t ldsA[2][64 * LDS_STRIDE];   // 18432 B
    __shared__ bf16_t ldsB[2][64 * LDS_STRIDE];   // 18432 B

    const int tid  = threadIdx.x;      // 0..127
    const int lane = tid & 31;
    const int wn   = tid >> 5;         // wave id 0..3 = N sub-block

    const int tileM = blockIdx.x / tilesN;
    const int tileN = blockIdx.x - tileM * tilesN;
    const int mBase = tileM * 64;
    const int nBase = tileN * 64;

    // ---- global->register staging mapping (16B chunks) ----
    // A tile: 64 rows x 8 chunks = 512 chunks; thread t -> chunks t + 128*i
    // (rows t>>3 + 16*i).  B tile identical.
    const int sRow = tid >> 3;         // 0..15 (+16 per chunk step)
    const int sCol = tid & 7;          // 16B chunk within row

    const bf16_t* aGlob = A + (size_t)(mBase + sRow) * K + sCol * 8;
    const bf16_t* bGlob = W + (size_t)(nBase + sRow) * K + sCol * 8;

    bf16_t* aLds = &ldsA[0][0] + sRow * LDS_STRIDE + sCol * 8;
    bf16_t* bLds = &ldsB[0][0] + sRow * LDS_STRIDE + sCol * 8;

    // ---- per-wave fragment read setup ----
    const int row16 = lane & 15;
    const int kOffE = (lane < 16) ? 0 : 8;       // element offset within 32-group
    const bf16_t* aFragBase = &ldsA[0][0] + row16 * LDS_STRIDE;  // + mt*16 rows
    const bf16_t* bFragBase = &ldsB[0][0] + (wn * 16 + row16) * LDS_STRIDE;
    const size_t  bufStride = (size_t)64 * LDS_STRIDE;           // elements

    v8f acc[4] = {};
    int4 stA[4];
    int4 stB[4];

    const int NK = K / 64;

    // ---- prologue: fetch K-slab 0 and commit to buffer 0 ----
    #pragma unroll
    for (int i = 0; i < 4; ++i) {
        stA[i] = *(const int4*)(aGlob + (size_t)(16 * i) * K);
        stB[i] = *(const int4*)(bGlob + (size_t)(16 * i) * K);
    }
    #pragma unroll
    for (int i = 0; i < 4; ++i) {
        *(int4*)(aLds + (size_t)(16 * i) * LDS_STRIDE) = stA[i];
        *(int4*)(bLds + (size_t)(16 * i) * LDS_STRIDE) = stB[i];
    }
    __syncthreads();

    for (int kb = 0; kb < NK; ++kb) {
        const int  buf  = kb & 1;
        const bool more = (kb + 1) < NK;

        // prefetch next K-slab into registers (latency hidden behind WMMAs)
        if (more) {
            const size_t gOff = (size_t)(kb + 1) * 64;
            #pragma unroll
            for (int i = 0; i < 4; ++i) {
                stA[i] = *(const int4*)(aGlob + (size_t)(16 * i) * K + gOff);
                stB[i] = *(const int4*)(bGlob + (size_t)(16 * i) * K + gOff);
            }
        }

        // compute: 2 sub-steps of K=32, 4 WMMAs each
        const bf16_t* aB = aFragBase + buf * bufStride;
        const bf16_t* bB = bFragBase + buf * bufStride;
        #pragma unroll
        for (int s = 0; s < 2; ++s) {
            const int sk = s * 32 + kOffE;       // element offset within row
            v16bf bfrag = frag_from_row(bB + sk);
            #pragma unroll
            for (int mt = 0; mt < 4; ++mt) {
                v16bf afrag = frag_from_row(aB + mt * 16 * LDS_STRIDE + sk);
                acc[mt] = __builtin_amdgcn_wmma_f32_16x16x32_bf16(
                    false, afrag, false, bfrag, (short)0, acc[mt], false, false);
            }
        }

        // commit prefetched slab to the other buffer
        if (more) {
            __syncthreads();   // all waves done reading buf^1 (from kb-1)
            bf16_t* aL = aLds + (buf ^ 1) * bufStride;
            bf16_t* bL = bLds + (buf ^ 1) * bufStride;
            #pragma unroll
            for (int i = 0; i < 4; ++i) {
                *(int4*)(aL + (size_t)(16 * i) * LDS_STRIDE) = stA[i];
                *(int4*)(bL + (size_t)(16 * i) * LDS_STRIDE) = stB[i];
            }
            __syncthreads();   // stores visible before next compute
        }
    }

    // ---- epilogue ----
    // C/D layout: n = lane%16, m = mt*16 + (lane<16 ? v : v+8)
    const int   n    = nBase + wn * 16 + row16;
    const float bn   = bias[n];
    const int   mOff = (lane < 16) ? 0 : 8;

    #pragma unroll
    for (int mt = 0; mt < 4; ++mt) {
        #pragma unroll
        for (int v = 0; v < 8; ++v) {
            const int m = mBase + mt * 16 + mOff + v;
            float val = acc[mt][v] + bn;
            if (APPLY_SIGMOID) val = sigmoidf_fast(val);
            C[(size_t)m * N + n] = val;
        }
    }
}

// ---------------------------------------------------------------------------
// GRU gates (h == 0 case): reads gi = x@w_ih^T + b_ih, produces h_new (bf16).
// ---------------------------------------------------------------------------
__global__ void gru_gate_kernel(const float* __restrict__ gi,   // [B, 3H]
                                const float* __restrict__ b_hh, // [3H]
                                bf16_t* __restrict__ h,         // [B, H]
                                int B, int H) {
    const int idx = blockIdx.x * blockDim.x + threadIdx.x;
    if (idx >= B * H) return;
    const int b = idx / H;
    const int j = idx - b * H;
    const float* row = gi + (size_t)b * 3 * H;
    const float r  = sigmoidf_fast(row[j]         + b_hh[j]);
    const float z  = sigmoidf_fast(row[H + j]     + b_hh[H + j]);
    const float nn = tanhf_fast   (row[2 * H + j] + r * b_hh[2 * H + j]);
    h[idx] = (bf16_t)((1.0f - z) * nn);
}

// ---------------------------------------------------------------------------
// Broadcast out2d [B,O] across T -> d_out [B,O,T]. Fully coalesced stores;
// consecutive T lanes read the same out2d word (L0 broadcast).
// ---------------------------------------------------------------------------
__global__ void broadcast_kernel(const float* __restrict__ out2d,
                                 float* __restrict__ out,
                                 unsigned T, size_t total) {
    size_t i = (size_t)blockIdx.x * blockDim.x + threadIdx.x;
    if (i < total) out[i] = out2d[i / T];
}

// ---------------------------------------------------------------------------
extern "C" void kernel_launch(void* const* d_in, const int* in_sizes, int n_in,
                              void* d_out, int out_size, void* d_ws, size_t ws_size,
                              hipStream_t stream) {
    const float* x     = (const float*)d_in[0];
    const float* w_ih  = (const float*)d_in[1];
    // d_in[2] (w_hh) is dead: h == 0 in every timestep of the reference.
    const float* b_ih  = (const float*)d_in[3];
    const float* b_hh  = (const float*)d_in[4];
    const float* w_cls = (const float*)d_in[5];
    const float* b_cls = (const float*)d_in[6];

    const int H = in_sizes[3] / 3;            // 2048
    const int I = in_sizes[1] / (3 * H);      // 2048
    const int B = in_sizes[0] / I;            // 1024
    const int O = in_sizes[5] / H;            // 2048
    const int T = out_size / (B * O);         // 30

    // Workspace layout (16B-aligned; ~64 MiB). out2d aliases gi (gi is dead
    // before the classifier GEMM runs; single-stream ordering keeps replays
    // deterministic).
    char* ws = (char*)d_ws;
    size_t off = 0;
    bf16_t* xb    = (bf16_t*)(ws + off); off += (size_t)B * I * sizeof(bf16_t);
    bf16_t* wihb  = (bf16_t*)(ws + off); off += (size_t)3 * H * I * sizeof(bf16_t);
    bf16_t* wclsb = (bf16_t*)(ws + off); off += (size_t)O * H * sizeof(bf16_t);
    bf16_t* hb    = (bf16_t*)(ws + off); off += (size_t)B * H * sizeof(bf16_t);
    float*  gi    = (float*) (ws + off); // [B, 3H]  (24 MiB)
    float*  out2d = gi;                  // [B, O]   (8 MiB, reuses gi region)

    // 1) downcast operands to bf16 (vectorized x4)
    {
        size_t n0 = (size_t)B * I / 4;
        f32_to_bf16_kernel<<<(unsigned)((n0 + 255) / 256), 256, 0, stream>>>(x, xb, n0);
        size_t n1 = (size_t)3 * H * I / 4;
        f32_to_bf16_kernel<<<(unsigned)((n1 + 255) / 256), 256, 0, stream>>>(w_ih, wihb, n1);
        size_t n2 = (size_t)O * H / 4;
        f32_to_bf16_kernel<<<(unsigned)((n2 + 255) / 256), 256, 0, stream>>>(w_cls, wclsb, n2);
    }

    // 2) GEMM1: gi = x @ w_ih^T + b_ih   [B x 3H]
    {
        const int tilesN = (3 * H) / 64;                 // 96
        const int blocks = (B / 64) * tilesN;            // 1536
        gemm_bf16_lds_kernel<false><<<blocks, 128, 0, stream>>>(
            xb, wihb, b_ih, gi, B, I, 3 * H, tilesN);
    }

    // 3) GRU gates -> h (bf16)
    {
        const int n = B * H;
        gru_gate_kernel<<<(n + 255) / 256, 256, 0, stream>>>(gi, b_hh, hb, B, H);
    }

    // 4) GEMM2: out2d = sigmoid(h @ w_cls^T + b_cls)   [B x O]
    {
        const int tilesN = O / 64;                       // 32
        const int blocks = (B / 64) * tilesN;            // 512
        gemm_bf16_lds_kernel<true><<<blocks, 128, 0, stream>>>(
            hb, wclsb, b_cls, out2d, B, H, O, tilesN);
    }

    // 5) broadcast across T -> d_out [B, O, T]
    {
        size_t total = (size_t)out_size;
        broadcast_kernel<<<(unsigned)((total + 255) / 256), 256, 0, stream>>>(
            out2d, (float*)d_out, (unsigned)T, total);
    }
}
